// MLPTexture3D_24421184045515
// MI455X (gfx1250) — compile-verified, matrix operand order checked by reading, not run on hardware
//
#include <hip/hip_runtime.h>
#include <math.h>

typedef __attribute__((ext_vector_type(16))) _Float16 v16h;
typedef __attribute__((ext_vector_type(8)))  _Float16 v8h;
typedef __attribute__((ext_vector_type(8)))  float    v8f;

#define NLVL 16
#define TBL  (1u << 19)

static __device__ __forceinline__ unsigned umin32(unsigned a, unsigned b) { return a < b ? a : b; }

// Build a 16x32 f16 B-fragment (K=32 x N=16) from row-major W[nrows][32],
// taking output columns [cbase, cbase+16). B[k][n] = W[n][k].
// Lane layout (wave32): lanes 0-15 -> col n, K octets {0..7, 16..23};
//                       lanes 16-31 -> col n, K octets {8..15, 24..31}.
static __device__ __forceinline__ v16h load_b_frag(const float* __restrict__ W,
                                                   int lane, int cbase, int nrows) {
  const int n  = (lane & 15) + cbase;
  const int k0 = (lane >> 4) ? 8  : 0;
  const int k1 = (lane >> 4) ? 24 : 16;
  v16h b;
#pragma unroll
  for (int j = 0; j < 16; ++j) b[j] = (_Float16)0.0f;
  if (n < nrows) {
    const float* r = W + n * 32;
#pragma unroll
    for (int j = 0; j < 8; ++j) {
      b[j]     = (_Float16)r[k0 + j];
      b[8 + j] = (_Float16)r[k1 + j];
    }
  }
  return b;
}

// Read 16x32 f16 A-fragment from a row-major [16][32] f16 LDS tile.
static __device__ __forceinline__ v16h load_a_frag(const _Float16* sb, int lane) {
  const int row = lane & 15;
  const int k0  = (lane >> 4) ? 8  : 0;
  const int k1  = (lane >> 4) ? 24 : 16;
  const v8h lo = *(const v8h*)(sb + row * 32 + k0);
  const v8h hi = *(const v8h*)(sb + row * 32 + k1);
  v16h a;
#pragma unroll
  for (int j = 0; j < 8; ++j) { a[j] = lo[j]; a[8 + j] = hi[j]; }
  return a;
}

// Write f32 16x16 D accumulator (ReLU'd) back into LDS tile columns [cbase,cbase+16).
// D layout: lanes 0-15: VGPR r -> (M=r, N=lane); lanes 16-31: VGPR r -> (M=r+8, N=lane-16).
static __device__ __forceinline__ void store_relu(_Float16* sb, int lane, int cbase, v8f d) {
  const int col = (lane & 15) + cbase;
  const int r0  = (lane >> 4) * 8;
#pragma unroll
  for (int r = 0; r < 8; ++r) {
    float v = d[r] > 0.0f ? d[r] : 0.0f;
    sb[(r0 + r) * 32 + col] = (_Float16)v;
  }
}

__global__ __launch_bounds__(256) void mlptex3d_kernel(
    const float* __restrict__ texc, const float* __restrict__ grid,
    const float* __restrict__ W0,   const float* __restrict__ W1,
    const float* __restrict__ W2,   const float* __restrict__ aabb,
    const float* __restrict__ minmax, float* __restrict__ out, int npts) {
  __shared__ __align__(16) _Float16 sbuf[8][16 * 32];  // per-wave activation tile
  __shared__ __align__(16) float    suvw[8][16 * 4];   // per-wave normalized coords

  const int lane  = threadIdx.x & 31;
  const int wid   = threadIdx.x >> 5;
  const int pbase = ((int)blockIdx.x * 8 + wid) * 16;

  _Float16* sb = sbuf[wid];
  float*    uv = suvw[wid];

  // ---- normalize coordinates into per-wave LDS (lanes 0-15, one point each) ----
  const float lox = aabb[0], loy = aabb[1], loz = aabb[2];
  const float ivx = 1.0f / (aabb[3] - aabb[0]);
  const float ivy = 1.0f / (aabb[4] - aabb[1]);
  const float ivz = 1.0f / (aabb[5] - aabb[2]);
  if (lane < 16) {
    int p = pbase + lane; if (p > npts - 1) p = npts - 1;
    const float* t = texc + (size_t)p * 3;
    float ux = fminf(fmaxf((t[0] - lox) * ivx, 0.0f), 1.0f);
    float uy = fminf(fmaxf((t[1] - loy) * ivy, 0.0f), 1.0f);
    float uz = fminf(fmaxf((t[2] - loz) * ivz, 0.0f), 1.0f);
    uv[lane * 4 + 0] = ux; uv[lane * 4 + 1] = uy; uv[lane * 4 + 2] = uz;
  }
  // same-wave LDS ops are in-order; no barrier needed (tile is wave-private)

  // ---- hash-grid encode: 8 iterations x 32 lanes = 256 (point,level) tasks ----
#pragma unroll
  for (int it = 0; it < 8; ++it) {
    const int level = it * 2 + (lane >> 4);
    const int pt    = lane & 15;
    const float ux = uv[pt * 4 + 0], uy = uv[pt * 4 + 1], uz = uv[pt * 4 + 2];

    // scale = 16 * (4096/16)^(l/15) - 1 = 16 * 2^(8l/15) - 1
    const float scale = exp2f((float)level * (8.0f / 15.0f)) * 16.0f - 1.0f;
    const unsigned res = (unsigned)(ceilf(scale) + 1.0f);
    const bool dense = ((unsigned long long)res * res * res) <= (unsigned long long)TBL;

    float px = ux * scale + 0.5f, py = uy * scale + 0.5f, pz = uz * scale + 0.5f;
    float fx0 = floorf(px), fy0 = floorf(py), fz0 = floorf(pz);
    float fx = px - fx0, fy = py - fy0, fz = pz - fz0;
    unsigned x0 = (unsigned)fx0, y0 = (unsigned)fy0, z0 = (unsigned)fz0;

    const float* glvl = grid + (size_t)level * (size_t)TBL * 2;
    float f0 = 0.0f, f1 = 0.0f;
#pragma unroll
    for (int c = 0; c < 8; ++c) {
      const unsigned ox = c & 1, oy = (c >> 1) & 1, oz = (c >> 2) & 1;
      unsigned cx = x0 + ox, cy = y0 + oy, cz = z0 + oz;
      // dense index (clamped) vs spatial hash; branchless select
      unsigned dx = umin32(cx, res - 1), dy = umin32(cy, res - 1), dz = umin32(cz, res - 1);
      unsigned idx_d = dx + dy * res + dz * res * res;
      unsigned idx_h = (cx * 1u) ^ (cy * 2654435761u) ^ (cz * 805459861u);
      idx_h &= (TBL - 1u);
      unsigned idx = dense ? idx_d : idx_h;
      float w = (ox ? fx : 1.0f - fx) * (oy ? fy : 1.0f - fy) * (oz ? fz : 1.0f - fz);
      float g0 = glvl[(size_t)idx * 2 + 0];
      float g1 = glvl[(size_t)idx * 2 + 1];
      f0 = fmaf(w, g0, f0);
      f1 = fmaf(w, g1, f1);
    }
    sb[pt * 32 + 2 * level + 0] = (_Float16)f0;
    sb[pt * 32 + 2 * level + 1] = (_Float16)f1;
  }

  // ---- weight fragments (B layout), f16 ----
  const v16h B00 = load_b_frag(W0, lane, 0,  32);
  const v16h B01 = load_b_frag(W0, lane, 16, 32);
  const v16h B10 = load_b_frag(W1, lane, 0,  32);
  const v16h B11 = load_b_frag(W1, lane, 16, 32);
  const v16h B2  = load_b_frag(W2, lane, 0,  9);

  const v8f zero = {};

  // ---- layer 0: enc[16x32] @ W0^T -> h[16x32], ReLU ----
  v16h a0 = load_a_frag(sb, lane);
  v8f h0 = __builtin_amdgcn_wmma_f32_16x16x32_f16(false, a0, false, B00, (short)0, zero, false, false);
  v8f h1 = __builtin_amdgcn_wmma_f32_16x16x32_f16(false, a0, false, B01, (short)0, zero, false, false);
  store_relu(sb, lane, 0,  h0);
  store_relu(sb, lane, 16, h1);

  // ---- layer 1 ----
  v16h a1 = load_a_frag(sb, lane);
  v8f g0 = __builtin_amdgcn_wmma_f32_16x16x32_f16(false, a1, false, B10, (short)0, zero, false, false);
  v8f g1 = __builtin_amdgcn_wmma_f32_16x16x32_f16(false, a1, false, B11, (short)0, zero, false, false);
  store_relu(sb, lane, 0,  g0);
  store_relu(sb, lane, 16, g1);

  // ---- layer 2: [16x32] @ W2^T (9 valid cols) ----
  v16h a2 = load_a_frag(sb, lane);
  v8f o = __builtin_amdgcn_wmma_f32_16x16x32_f16(false, a2, false, B2, (short)0, zero, false, false);

  // ---- sigmoid * (max-min) + min, scatter to out[N][9] ----
  const int col = lane & 15;
  const int r0  = (lane >> 4) * 8;
  if (col < 9) {
    const float mlo = minmax[col];
    const float mrange = minmax[9 + col] - mlo;
#pragma unroll
    for (int r = 0; r < 8; ++r) {
      const int p = pbase + r0 + r;
      if (p < npts) {
        float s = 1.0f / (1.0f + __expf(-o[r]));
        out[(size_t)p * 9 + col] = fmaf(s, mrange, mlo);
      }
    }
  }
}

extern "C" void kernel_launch(void* const* d_in, const int* in_sizes, int n_in,
                              void* d_out, int out_size, void* d_ws, size_t ws_size,
                              hipStream_t stream) {
  const float* texc   = (const float*)d_in[0];
  const float* grid   = (const float*)d_in[1];
  const float* W0     = (const float*)d_in[2];
  const float* W1     = (const float*)d_in[3];
  const float* W2     = (const float*)d_in[4];
  const float* aabb   = (const float*)d_in[5];
  const float* minmax = (const float*)d_in[6];
  float* out = (float*)d_out;

  const int npts = in_sizes[0] / 3;
  const int blocks = (npts + 127) / 128;  // 8 waves x 16 points per block
  mlptex3d_kernel<<<blocks, 256, 0, stream>>>(texc, grid, W0, W1, W2, aabb, minmax, out, npts);
}
